// SpectroTFDecoder_27410481283735
// MI455X (gfx1250) — compile-verified, matrix operand order checked by reading, not run on hardware
//
#include <hip/hip_runtime.h>
#include <math.h>
#include <stdint.h>

// ---------------------------------------------------------------------------
// SpectroTFDecoder for MI455X (gfx1250): bf16 WMMA pipeline.
//   B=4, T=1024, D=1024, H=16, DH=64, DFF=4096. Compute-bound at bf16
//   (~172 GFLOP vs ~150MB HBM) -> v_wmma_f32_16x16x32_bf16 everywhere,
//   flash-fused attention (scores never leave LDS), TDM async A-tile staging.
// ---------------------------------------------------------------------------

typedef __bf16 bf16_t;
typedef __bf16 v16bf __attribute__((ext_vector_type(16)));
typedef __bf16 v8bf  __attribute__((ext_vector_type(8)));
typedef float  v8f   __attribute__((ext_vector_type(8)));
typedef float  v4f   __attribute__((ext_vector_type(4)));
typedef unsigned int v4u __attribute__((ext_vector_type(4)));
typedef int    v8i_  __attribute__((ext_vector_type(8)));
typedef int    v4i_  __attribute__((ext_vector_type(4)));

#define D_MODEL 1024
#define DFF_    4096
#define NH      16
#define DH_     64
#define BATCH   4
#define TQ_     1024
#define TK_     1024
#define MROWS   (BATCH * TQ_)   // 4096
#define LN_EPS  1e-6f

#if __has_builtin(__builtin_amdgcn_tensor_load_to_lds) && \
    __has_builtin(__builtin_amdgcn_s_wait_tensorcnt)
#define HAVE_TDM 1
#else
#define HAVE_TDM 0
#endif

struct BFPair { v8bf lo, hi; };

static __device__ __forceinline__ v16bf mkfrag(v8bf lo, v8bf hi) {
    BFPair p{lo, hi};
    return __builtin_bit_cast(v16bf, p);
}

static __device__ __forceinline__ void wait_ds0() {
    asm volatile("s_wait_dscnt 0" ::: "memory");
}

#if HAVE_TDM
// Tensor Data Mover: DMA a [rows=128][cols=32] bf16 tile (row stride K elems)
// from global memory into contiguous LDS at ldsOff. D# layout per CDNA5 ISA
// ch.8: group0 = {count, lds_addr, global_addr[56:0], type=2},
// group1 = {data_size=2B, tensor_dim0=K, tensor_dim1=M, tile 32x128,
//           tensor_dim0_stride=K}. Tracked by TENSORcnt.
static __device__ __forceinline__ void tdm_load_tile_A(const bf16_t* gptr,
                                                       unsigned ldsOff,
                                                       int K, int M) {
    unsigned long long ga = (unsigned long long)(uintptr_t)gptr;
    v4u g0 = { 1u,                                  // count=1 valid descriptor
               ldsOff,                              // lds_addr (bytes)
               (unsigned)ga,                        // global_addr[31:0]
               (unsigned)(((ga >> 32) & 0x01FFFFFFull) | (2u << 30)) }; // +type=2
    v8i_ g1 = { (int)0x00010000u,                               // data_size=1 (2B)
                (int)(((unsigned)K & 0xFFFFu) << 16),           // tensor_dim0 lo
                (int)((((unsigned)K >> 16) & 0xFFFFu) |
                      (((unsigned)M & 0xFFFFu) << 16)),         // td0 hi | td1 lo
                (int)((((unsigned)M >> 16) & 0xFFFFu) |
                      (32u << 16)),                             // td1 hi | tile_dim0=32
                (int)128,                                       // tile_dim1=128 rows
                (int)K,                                         // dim0_stride lo
                0, 0 };                                         // stride hi, dim1_stride
    v4i_ z4 = {0, 0, 0, 0};
#if __clang_major__ >= 23
    v8i_ z8 = {0, 0, 0, 0, 0, 0, 0, 0};
    __builtin_amdgcn_tensor_load_to_lds(g0, g1, z4, z4, z8, 0);
#else
    __builtin_amdgcn_tensor_load_to_lds(g0, g1, z4, z4, 0);
#endif
}
#endif

// ---------------------------------------------------------------------------
// fp32 -> bf16 elementwise convert (weights, src_x)
// ---------------------------------------------------------------------------
__global__ void f32_to_bf16_kernel(const float* __restrict__ in,
                                   bf16_t* __restrict__ out, int n) {
    int i = blockIdx.x * blockDim.x + threadIdx.x;
    int stride = gridDim.x * blockDim.x;
    for (; i < n; i += stride) out[i] = (bf16_t)in[i];
}

// ---------------------------------------------------------------------------
// LayerNorm: one 256-thread block per row of D=1024. eps added to sigma.
// ---------------------------------------------------------------------------
__global__ __launch_bounds__(256)
void ln_kernel(const float* __restrict__ x, const float* __restrict__ g,
               const float* __restrict__ b, bf16_t* __restrict__ out) {
    __shared__ float red[256];
    const int row = blockIdx.x;
    const int t = threadIdx.x;
    const float* xr = x + (size_t)row * D_MODEL;
    bf16_t* outr = out + (size_t)row * D_MODEL;

    v4f val = *(const v4f*)(xr + t * 4);
    red[t] = val[0] + val[1] + val[2] + val[3];
    __syncthreads();
    for (int k = 128; k > 0; k >>= 1) {
        if (t < k) red[t] += red[t + k];
        __syncthreads();
    }
    const float mu = red[0] * (1.0f / D_MODEL);
    __syncthreads();

    float d0 = val[0] - mu, d1 = val[1] - mu, d2 = val[2] - mu, d3 = val[3] - mu;
    red[t] = d0 * d0 + d1 * d1 + d2 * d2 + d3 * d3;
    __syncthreads();
    for (int k = 128; k > 0; k >>= 1) {
        if (t < k) red[t] += red[t + k];
        __syncthreads();
    }
    const float sigma = sqrtf(red[0] * (1.0f / D_MODEL));
    const float inv = 1.0f / (sigma + LN_EPS);

    #pragma unroll
    for (int j = 0; j < 4; ++j) {
        int c = t * 4 + j;
        outr[c] = (bf16_t)((val[j] - mu) * inv * g[c] + b[c]);
    }
}

// ---------------------------------------------------------------------------
// Tiled bf16 GEMM: C[M,N] = A[M,K] x B[K,N] + bias (+resid)(+relu)
// Block: 256 thr = 8 waves; tile 128(M) x 128(N); BK=32 double-buffered.
// Wave = 32x64 C tile -> 8 WMMAs per 6 fragment loads per K-step.
// A tile staged by the Tensor Data Mover (async, TENSORcnt) when available;
// B staged transposed ([n][k]) by all threads.
// ---------------------------------------------------------------------------
template <int OUT_BF16, int RELU, int RESID>
__global__ __launch_bounds__(256)
void gemm_kernel(const bf16_t* __restrict__ A, const bf16_t* __restrict__ B,
                 const float* __restrict__ bias, const float* __restrict__ resid,
                 void* __restrict__ Cout, int M, int N, int K) {
    __shared__ bf16_t As[2][128 * 32];  // row-major [128][32]
    __shared__ bf16_t Bs[2][128 * 32];  // transposed [n][k]

    const int tid  = threadIdx.x;
    const int lane = tid & 31;
    const int wave = tid >> 5;
    const int wm   = wave & 3;   // M sub-tile (0..3), 32 rows each
    const int wn   = wave >> 2;  // N sub-tile (0..1), 64 cols each
    const int bm   = blockIdx.y * 128;
    const int bn   = blockIdx.x * 128;
    const int half = lane >> 4;
    const int l15  = lane & 15;

    v8f acc[2][4] = {};

#if !HAVE_TDM
    auto stageA = [&](int buf, int k0) {
        #pragma unroll
        for (int i = 0; i < 2; ++i) {
            int c  = tid * 2 + i;      // 0..511 chunks of 8 bf16
            int r  = c >> 2;           // row 0..127
            int cc = (c & 3) * 8;
            *(v8bf*)(&As[buf][r * 32 + cc]) =
                *(const v8bf*)(A + (size_t)(bm + r) * K + k0 + cc);
        }
    };
#endif
    auto stageB = [&](int buf, int k0) {
        #pragma unroll
        for (int i = 0; i < 2; ++i) {
            int c  = tid * 2 + i;      // 0..511: 32 k-rows x 16 n-chunks
            int k  = c >> 4;
            int nb = (c & 15) * 8;
            v8bf vv = *(const v8bf*)(B + (size_t)(k0 + k) * N + bn + nb);
            #pragma unroll
            for (int j = 0; j < 8; ++j)
                Bs[buf][(nb + j) * 32 + k] = vv[j];   // transpose into LDS
        }
    };
    auto loadAfrag = [&](int buf, int mi) -> v16bf {
        const bf16_t* p = &As[buf][(wm * 32 + mi * 16 + l15) * 32];
        return mkfrag(*(const v8bf*)(p + half * 8),
                      *(const v8bf*)(p + 16 + half * 8));
    };
    auto loadBfrag = [&](int buf, int ni) -> v16bf {
        const bf16_t* p = &Bs[buf][(wn * 64 + ni * 16 + l15) * 32];
        return mkfrag(*(const v8bf*)(p + half * 8),
                      *(const v8bf*)(p + 16 + half * 8));
    };

    // ---- prologue: stage first K-tile ----
#if HAVE_TDM
    if (wave == 0)
        tdm_load_tile_A(A + (size_t)bm * K,
                        (unsigned)(uintptr_t)(void*)&As[0][0], K, M);
#else
    stageA(0, 0);
#endif
    stageB(0, 0);
#if HAVE_TDM
    if (wave == 0) __builtin_amdgcn_s_wait_tensorcnt(0);
#endif
    __syncthreads();

    for (int k0 = 0; k0 < K; k0 += 32) {
        const int cur = (k0 >> 5) & 1;
        if (k0 + 32 < K) {
#if HAVE_TDM
            if (wave == 0)
                tdm_load_tile_A(A + (size_t)bm * K + k0 + 32,
                                (unsigned)(uintptr_t)(void*)&As[cur ^ 1][0], K, M);
#else
            stageA(cur ^ 1, k0 + 32);
#endif
            stageB(cur ^ 1, k0 + 32);
            if (k0 + 64 < K)  // L2 prefetch of the next B tile (rows k0+32..k0+63)
                __builtin_prefetch(B + (size_t)(k0 + 32 + (tid >> 3)) * N + bn, 0, 1);
        }
        v16bf a0 = loadAfrag(cur, 0), a1 = loadAfrag(cur, 1);
        v16bf b0 = loadBfrag(cur, 0), b1 = loadBfrag(cur, 1);
        v16bf b2 = loadBfrag(cur, 2), b3 = loadBfrag(cur, 3);
        acc[0][0] = __builtin_amdgcn_wmma_f32_16x16x32_bf16(false, a0, false, b0, (short)0, acc[0][0], false, false);
        acc[0][1] = __builtin_amdgcn_wmma_f32_16x16x32_bf16(false, a0, false, b1, (short)0, acc[0][1], false, false);
        acc[0][2] = __builtin_amdgcn_wmma_f32_16x16x32_bf16(false, a0, false, b2, (short)0, acc[0][2], false, false);
        acc[0][3] = __builtin_amdgcn_wmma_f32_16x16x32_bf16(false, a0, false, b3, (short)0, acc[0][3], false, false);
        acc[1][0] = __builtin_amdgcn_wmma_f32_16x16x32_bf16(false, a1, false, b0, (short)0, acc[1][0], false, false);
        acc[1][1] = __builtin_amdgcn_wmma_f32_16x16x32_bf16(false, a1, false, b1, (short)0, acc[1][1], false, false);
        acc[1][2] = __builtin_amdgcn_wmma_f32_16x16x32_bf16(false, a1, false, b2, (short)0, acc[1][2], false, false);
        acc[1][3] = __builtin_amdgcn_wmma_f32_16x16x32_bf16(false, a1, false, b3, (short)0, acc[1][3], false, false);
#if HAVE_TDM
        if (wave == 0) __builtin_amdgcn_s_wait_tensorcnt(0);
#endif
        __syncthreads();
    }

    #pragma unroll
    for (int mi = 0; mi < 2; ++mi) {
        #pragma unroll
        for (int ni = 0; ni < 4; ++ni) {
            const int col = bn + wn * 64 + ni * 16 + l15;
            const float bv = bias[col];
            #pragma unroll
            for (int i = 0; i < 8; ++i) {
                const int row = bm + wm * 32 + mi * 16 + i + 8 * half;
                float v = acc[mi][ni][i] + bv;
                if (RELU)  v = v > 0.f ? v : 0.f;
                if (RESID) v += resid[(size_t)row * N + col];
                if (OUT_BF16) ((bf16_t*)Cout)[(size_t)row * N + col] = (bf16_t)v;
                else          ((float*)Cout)[(size_t)row * N + col] = v;
            }
        }
    }
}

// ---------------------------------------------------------------------------
// Flash attention (bf16 QKV, f32 accum). Block = 4 waves, 64 q-rows of one
// (b,h). S = Q K^T via WMMA (row-major K IS the B operand), online softmax in
// LDS, P.V via WMMA with V staged transposed. Scores never touch HBM
// (saves 2 x 268 MB of round-trip traffic = ~46 us at 23.3 TB/s).
// ---------------------------------------------------------------------------
__global__ __launch_bounds__(128)
void attn_kernel(const bf16_t* __restrict__ Q, const bf16_t* __restrict__ Km,
                 const bf16_t* __restrict__ V, bf16_t* __restrict__ O,
                 int Tq, int Tk, int causal) {
    __shared__ float  Sbuf[4][16][64];
    __shared__ bf16_t Pbuf[4][16][72];   // stride 72 keeps 16B row alignment
    __shared__ bf16_t Vt[64][72];        // transposed V tile [dh][key]
    __shared__ float  AlphaS[4][16];
    __shared__ float  LinvS[4][16];

    const int tid  = threadIdx.x;
    const int lane = tid & 31;
    const int wave = tid >> 5;
    const int half = lane >> 4;
    const int l15  = lane & 15;
    const int q0   = blockIdx.x * 64;
    const int h    = blockIdx.y;
    const int b    = blockIdx.z;

    const bf16_t* Qb = Q  + ((size_t)b * Tq) * D_MODEL + h * DH_;
    const bf16_t* Kb = Km + ((size_t)b * Tk) * D_MODEL + h * DH_;
    const bf16_t* Vb = V  + ((size_t)b * Tk) * D_MODEL + h * DH_;
    bf16_t*       Ob = O  + ((size_t)b * Tq) * D_MODEL + h * DH_;

    const int qrow = q0 + wave * 16 + l15;
    v16bf qa[2];
    #pragma unroll
    for (int s = 0; s < 2; ++s) {
        const bf16_t* p = Qb + (size_t)qrow * D_MODEL + s * 32;
        qa[s] = mkfrag(*(const v8bf*)(p + half * 8),
                       *(const v8bf*)(p + 16 + half * 8));
    }

    v8f acc[4] = {};
    float m_i = -INFINITY, l_i = 0.f;

    int ktEnd = causal ? (q0 + 64) : Tk;
    if (ktEnd > Tk) ktEnd = Tk;

    for (int kt = 0; kt < ktEnd; kt += 64) {
        // ---- stage V tile transposed (shared by all 4 waves) ----
        __syncthreads();
        #pragma unroll
        for (int i = 0; i < 4; ++i) {
            int c   = tid + i * 128;
            int key = c >> 3;
            int db  = (c & 7) * 8;
            v8bf vv = *(const v8bf*)(Vb + (size_t)(kt + key) * D_MODEL + db);
            #pragma unroll
            for (int j = 0; j < 8; ++j) Vt[db + j][key] = vv[j];
        }
        __syncthreads();

        // ---- S = Q K^T (this wave's 16 rows x 64 keys) ----
        #pragma unroll
        for (int ktile = 0; ktile < 4; ++ktile) {
            v8f s = {};
            const int key = kt + ktile * 16 + l15;
            #pragma unroll
            for (int st = 0; st < 2; ++st) {
                const bf16_t* p = Kb + (size_t)key * D_MODEL + st * 32;
                v16bf bfr = mkfrag(*(const v8bf*)(p + half * 8),
                                   *(const v8bf*)(p + 16 + half * 8));
                s = __builtin_amdgcn_wmma_f32_16x16x32_bf16(false, qa[st], false, bfr, (short)0, s, false, false);
            }
            #pragma unroll
            for (int i = 0; i < 8; ++i)
                Sbuf[wave][i + 8 * half][ktile * 16 + l15] = s[i];
        }
        wait_ds0();

        // ---- online softmax; lane r (<16) owns q-row r of this wave ----
        if (lane < 16) {
            const int rowg = q0 + wave * 16 + lane;
            float mx = m_i;
            for (int c = 0; c < 64; ++c) {
                float sv = Sbuf[wave][lane][c] * 0.125f;  // 1/sqrt(64)
                if (causal && (kt + c) > rowg) sv = -INFINITY;
                Sbuf[wave][lane][c] = sv;
                mx = fmaxf(mx, sv);
            }
            float alpha, sum = 0.f;
            if (mx == -INFINITY) {
                alpha = 1.f;
                for (int c = 0; c < 64; ++c) Pbuf[wave][lane][c] = (bf16_t)0.f;
            } else {
                alpha = __expf(m_i - mx);
                for (int c = 0; c < 64; ++c) {
                    float p = __expf(Sbuf[wave][lane][c] - mx);
                    sum += p;
                    Pbuf[wave][lane][c] = (bf16_t)p;
                }
            }
            l_i = l_i * alpha + sum;
            m_i = mx;
            AlphaS[wave][lane] = alpha;
        }
        wait_ds0();

        // ---- rescale O accumulators by alpha(row) ----
        float al[8];
        #pragma unroll
        for (int i = 0; i < 8; ++i) al[i] = AlphaS[wave][i + 8 * half];
        #pragma unroll
        for (int ni = 0; ni < 4; ++ni)
            #pragma unroll
            for (int i = 0; i < 8; ++i) acc[ni][i] *= al[i];

        // ---- O += P (16x64) x V (64x64) ----
        #pragma unroll
        for (int st = 0; st < 2; ++st) {
            const bf16_t* pa = &Pbuf[wave][l15][st * 32];
            v16bf afr = mkfrag(*(const v8bf*)(pa + half * 8),
                               *(const v8bf*)(pa + 16 + half * 8));
            #pragma unroll
            for (int ni = 0; ni < 4; ++ni) {
                const bf16_t* pb = &Vt[ni * 16 + l15][st * 32];
                v16bf bfr = mkfrag(*(const v8bf*)(pb + half * 8),
                                   *(const v8bf*)(pb + 16 + half * 8));
                acc[ni] = __builtin_amdgcn_wmma_f32_16x16x32_bf16(false, afr, false, bfr, (short)0, acc[ni], false, false);
            }
        }
    }

    if (lane < 16) LinvS[wave][lane] = (l_i > 0.f) ? (1.f / l_i) : 0.f;
    wait_ds0();
    float li[8];
    #pragma unroll
    for (int i = 0; i < 8; ++i) li[i] = LinvS[wave][i + 8 * half];
    #pragma unroll
    for (int ni = 0; ni < 4; ++ni)
        #pragma unroll
        for (int i = 0; i < 8; ++i) {
            const int row = q0 + wave * 16 + i + 8 * half;
            Ob[(size_t)row * D_MODEL + ni * 16 + l15] = (bf16_t)(acc[ni][i] * li[i]);
        }
}

// ---------------------------------------------------------------------------
// Host pipeline
// ---------------------------------------------------------------------------
extern "C" void kernel_launch(void* const* d_in, const int* in_sizes, int n_in,
                              void* d_out, int out_size, void* d_ws, size_t ws_size,
                              hipStream_t stream) {
    (void)in_sizes; (void)n_in; (void)out_size; (void)ws_size;

    const float* x     = (const float*)d_in[0];
    const float* src_x = (const float*)d_in[1];
    // d_in[2] mask = causal(TQ x TQ), d_in[3] src_mask = all-false -> flags
    const float* ln_g[3] = {(const float*)d_in[4], (const float*)d_in[6], (const float*)d_in[8]};
    const float* ln_b[3] = {(const float*)d_in[5], (const float*)d_in[7], (const float*)d_in[9]};
    const float* sa_w[4] = {(const float*)d_in[10], (const float*)d_in[11],
                            (const float*)d_in[12], (const float*)d_in[13]};
    const float* ca_w[4] = {(const float*)d_in[14], (const float*)d_in[15],
                            (const float*)d_in[16], (const float*)d_in[17]};
    const float* sa_b[4] = {(const float*)d_in[18], (const float*)d_in[19],
                            (const float*)d_in[20], (const float*)d_in[21]};
    const float* ca_b[4] = {(const float*)d_in[22], (const float*)d_in[23],
                            (const float*)d_in[24], (const float*)d_in[25]};
    const float* ff_w1 = (const float*)d_in[26];
    const float* ff_b1 = (const float*)d_in[27];
    const float* ff_w2 = (const float*)d_in[28];
    const float* ff_b2 = (const float*)d_in[29];
    float* out = (float*)d_out;

    // ---- workspace carve-up (~128 MB) ----
    size_t off = 0;
    auto alloc = [&](size_t bytes) -> void* {
        off = (off + 255) & ~(size_t)255;
        void* p = (char*)d_ws + off;
        off += bytes;
        return p;
    };
    bf16_t* wsa[4]; bf16_t* wca[4];
    for (int i = 0; i < 4; ++i) wsa[i] = (bf16_t*)alloc((size_t)D_MODEL * D_MODEL * 2);
    for (int i = 0; i < 4; ++i) wca[i] = (bf16_t*)alloc((size_t)D_MODEL * D_MODEL * 2);
    bf16_t* wff1 = (bf16_t*)alloc((size_t)D_MODEL * DFF_ * 2);
    bf16_t* wff2 = (bf16_t*)alloc((size_t)DFF_ * D_MODEL * 2);
    bf16_t* hb   = (bf16_t*)alloc((size_t)MROWS * D_MODEL * 2);
    bf16_t* srcb = (bf16_t*)alloc((size_t)MROWS * D_MODEL * 2);
    bf16_t* qb   = (bf16_t*)alloc((size_t)MROWS * D_MODEL * 2);
    bf16_t* kb   = (bf16_t*)alloc((size_t)MROWS * D_MODEL * 2);
    bf16_t* vb   = (bf16_t*)alloc((size_t)MROWS * D_MODEL * 2);
    bf16_t* ab   = (bf16_t*)alloc((size_t)MROWS * D_MODEL * 2);
    bf16_t* ff1b = (bf16_t*)alloc((size_t)MROWS * DFF_ * 2);
    float*  x1   = (float*)alloc((size_t)MROWS * D_MODEL * 4);
    // x2 lives in d_out (written once, then read back as the final residual)

    const int DD = D_MODEL * D_MODEL;
    const int DF = D_MODEL * DFF_;

    for (int i = 0; i < 4; ++i)
        f32_to_bf16_kernel<<<1024, 256, 0, stream>>>(sa_w[i], wsa[i], DD);
    for (int i = 0; i < 4; ++i)
        f32_to_bf16_kernel<<<1024, 256, 0, stream>>>(ca_w[i], wca[i], DD);
    f32_to_bf16_kernel<<<1024, 256, 0, stream>>>(ff_w1, wff1, DF);
    f32_to_bf16_kernel<<<1024, 256, 0, stream>>>(ff_w2, wff2, DF);
    f32_to_bf16_kernel<<<1024, 256, 0, stream>>>(src_x, srcb, MROWS * D_MODEL);

    const dim3 gD(D_MODEL / 128, MROWS / 128);  // GEMMs with N=1024
    const dim3 gF(DFF_ / 128,    MROWS / 128);  // GEMM with N=4096
    const dim3 gA(TQ_ / 64, NH, BATCH);         // attention

    // ---- self-attention block ----
    ln_kernel<<<MROWS, 256, 0, stream>>>(x, ln_g[0], ln_b[0], hb);
    gemm_kernel<1, 0, 0><<<gD, 256, 0, stream>>>(hb, wsa[0], sa_b[0], nullptr, qb, MROWS, D_MODEL, D_MODEL);
    gemm_kernel<1, 0, 0><<<gD, 256, 0, stream>>>(hb, wsa[1], sa_b[1], nullptr, kb, MROWS, D_MODEL, D_MODEL);
    gemm_kernel<1, 0, 0><<<gD, 256, 0, stream>>>(hb, wsa[2], sa_b[2], nullptr, vb, MROWS, D_MODEL, D_MODEL);
    attn_kernel<<<gA, 128, 0, stream>>>(qb, kb, vb, ab, TQ_, TQ_, /*causal=*/1);
    gemm_kernel<0, 0, 1><<<gD, 256, 0, stream>>>(ab, wsa[3], sa_b[3], x, x1, MROWS, D_MODEL, D_MODEL);

    // ---- cross-attention block (K/V from raw src_x per reference) ----
    ln_kernel<<<MROWS, 256, 0, stream>>>(x1, ln_g[1], ln_b[1], hb);
    gemm_kernel<1, 0, 0><<<gD, 256, 0, stream>>>(hb,   wca[0], ca_b[0], nullptr, qb, MROWS, D_MODEL, D_MODEL);
    gemm_kernel<1, 0, 0><<<gD, 256, 0, stream>>>(srcb, wca[1], ca_b[1], nullptr, kb, MROWS, D_MODEL, D_MODEL);
    gemm_kernel<1, 0, 0><<<gD, 256, 0, stream>>>(srcb, wca[2], ca_b[2], nullptr, vb, MROWS, D_MODEL, D_MODEL);
    attn_kernel<<<gA, 128, 0, stream>>>(qb, kb, vb, ab, TQ_, TK_, /*causal=*/0);
    gemm_kernel<0, 0, 1><<<gD, 256, 0, stream>>>(ab, wca[3], ca_b[3], x1, out, MROWS, D_MODEL, D_MODEL); // x2 -> d_out

    // ---- feed-forward block (residual x2 read back from d_out) ----
    ln_kernel<<<MROWS, 256, 0, stream>>>(out, ln_g[2], ln_b[2], hb);
    gemm_kernel<1, 1, 0><<<gF, 256, 0, stream>>>(hb, wff1, ff_b1, nullptr, ff1b, MROWS, DFF_, D_MODEL);
    gemm_kernel<0, 0, 1><<<gD, 256, 0, stream>>>(ff1b, wff2, ff_b2, out, out, MROWS, D_MODEL, DFF_);
}
